// DATransformerBlock_29652454211864
// MI455X (gfx1250) — compile-verified, matrix operand order checked by reading, not run on hardware
//
#include <hip/hip_runtime.h>

typedef __attribute__((ext_vector_type(16))) _Float16 v16h;
typedef __attribute__((ext_vector_type(8)))  float    v8f;

namespace {
constexpr int Bc  = 2;
constexpr int Hc  = 32;
constexpr int Wc  = 32;
constexpr int Cc  = 384;
constexpr int NHc = 12;
constexpr int NGc = 6;
constexpr int GHc = 2;          // NH / NG
constexpr int CGc = 64;         // C / NG
constexpr int HCc = 32;         // C / NH
constexpr int HIDc = 1536;      // 4*C
constexpr int Lc  = 1024;       // H*W
constexpr int NSc = 1024;       // sampled points
constexpr int BHc = Bc * NHc;   // 24
constexpr int BGc = Bc * NGc;   // 12
constexpr float EPSc = 1e-5f;
}

__device__ __forceinline__ float gelu_f(float x) {
  return 0.5f * x * (1.0f + erff(x * 0.7071067811865475f));
}

// ---------------------------------------------------------------------------
// WMMA fragment load: 16-bit A/B matrix, 16x32 tile, per gfx1250 layout.
// lanes 0-15: M(or N)=lane, K halves {half*8+2i, half*8+2i+1} and +16 block.
// Pairs are 4-byte aligned (even k, even ld); compiler fuses these into
// global_load_b128 pairs (verified in round-1 disassembly).
// ---------------------------------------------------------------------------
__device__ __forceinline__ v16h load_frag16(const _Float16* __restrict__ p, int half) {
  union { v16h h; unsigned int u[8]; } f;
#pragma unroll
  for (int i = 0; i < 4; ++i) {
    f.u[i]     = *reinterpret_cast<const unsigned int*>(p + half * 8 + 2 * i);
    f.u[4 + i] = *reinterpret_cast<const unsigned int*>(p + 16 + half * 8 + 2 * i);
  }
  return f.h;
}

__device__ __forceinline__ v8f wmma_f16(v16h a, v16h b, v8f c) {
  return __builtin_amdgcn_wmma_f32_16x16x32_f16(
      /*neg_a=*/false, a, /*neg_b=*/false, b,
      /*c_mod=*/(short)0, c, /*reuse_a=*/false, /*reuse_b=*/false);
}

// ---------------------------------------------------------------------------
// Generic batched NT GEMM, 2x2 register-blocked:
//   C[z][m][n] = alpha * sum_k A[z][m][k]*Bt[z][n][k] + bias[n]
// One wave computes a 32x32 output tile: 2 A-frags x 2 B-frags -> 4 WMMAs per
// 32-wide k-step (16 flop/byte of fragment traffic). M,N multiples of 32,
// K multiple of 32. Optional f32/f16 outputs; act==1 applies exact GELU.
// ---------------------------------------------------------------------------
__global__ void wmma_gemm_nt(const _Float16* __restrict__ A,  int lda, long long sA,
                             const _Float16* __restrict__ Bt, int ldb, long long sB,
                             float* __restrict__ Cf, _Float16* __restrict__ Ch,
                             int ldc, long long sC,
                             const float* __restrict__ bias,
                             int M, int N, int K, float alpha, int act)
{
  const int zb = blockIdx.z;
  const _Float16* Ab = A  + zb * sA;
  const _Float16* Bb = Bt + zb * sB;
  const int ntiles = N >> 5;                               // 32-wide tiles
  const int mtiles = M >> 5;
  const int tile = blockIdx.x * blockDim.y + threadIdx.y;  // wave-uniform
  if (tile >= mtiles * ntiles) return;                     // whole-wave exit
  const int tm = tile / ntiles;
  const int tn = tile - tm * ntiles;
  const int lane = threadIdx.x;
  const int r  = lane & 15;
  const int hf = lane >> 4;
  const _Float16* Arow0 = Ab + (long long)(tm * 32 + r) * lda;
  const _Float16* Arow1 = Arow0 + (long long)16 * lda;
  const _Float16* Brow0 = Bb + (long long)(tn * 32 + r) * ldb;
  const _Float16* Brow1 = Brow0 + (long long)16 * ldb;

  v8f acc00 = {0.f,0.f,0.f,0.f,0.f,0.f,0.f,0.f};
  v8f acc01 = {0.f,0.f,0.f,0.f,0.f,0.f,0.f,0.f};
  v8f acc10 = {0.f,0.f,0.f,0.f,0.f,0.f,0.f,0.f};
  v8f acc11 = {0.f,0.f,0.f,0.f,0.f,0.f,0.f,0.f};
  for (int k = 0; k < K; k += 32) {
    // prefetch one k-block ahead (lowers to global_prefetch_b8; speculative
    // past-the-end prefetch is dropped by hardware)
    __builtin_prefetch(Arow0 + k + 32, 0, 1);
    __builtin_prefetch(Brow0 + k + 32, 0, 1);
    v16h a0 = load_frag16(Arow0 + k, hf);
    v16h a1 = load_frag16(Arow1 + k, hf);
    v16h b0 = load_frag16(Brow0 + k, hf);
    v16h b1 = load_frag16(Brow1 + k, hf);
    acc00 = wmma_f16(a0, b0, acc00);
    acc01 = wmma_f16(a0, b1, acc01);
    acc10 = wmma_f16(a1, b0, acc10);
    acc11 = wmma_f16(a1, b1, acc11);
  }

  const int n0 = tn * 32 + r;
  const int n1 = n0 + 16;
  const float bv0 = bias ? bias[n0] : 0.0f;
  const float bv1 = bias ? bias[n1] : 0.0f;
#pragma unroll
  for (int i = 0; i < 8; ++i) {
    const int m0 = tm * 32 + hf * 8 + i;   // C/D layout: M = vgpr + 8*(lane/16)
    const int m1 = m0 + 16;
    float v00 = acc00[i] * alpha + bv0;
    float v01 = acc01[i] * alpha + bv1;
    float v10 = acc10[i] * alpha + bv0;
    float v11 = acc11[i] * alpha + bv1;
    if (act == 1) { v00 = gelu_f(v00); v01 = gelu_f(v01); v10 = gelu_f(v10); v11 = gelu_f(v11); }
    const long long o00 = zb * sC + (long long)m0 * ldc + n0;
    const long long o01 = zb * sC + (long long)m0 * ldc + n1;
    const long long o10 = zb * sC + (long long)m1 * ldc + n0;
    const long long o11 = zb * sC + (long long)m1 * ldc + n1;
    if (Cf) { Cf[o00] = v00; Cf[o01] = v01; Cf[o10] = v10; Cf[o11] = v11; }
    if (Ch) { Ch[o00] = (_Float16)v00; Ch[o01] = (_Float16)v01;
              Ch[o10] = (_Float16)v10; Ch[o11] = (_Float16)v11; }
  }
}

// ---------------------------------------------------------------------------
// Elementwise / copy helpers
// ---------------------------------------------------------------------------
__global__ void to_half_k(const float* __restrict__ in, _Float16* __restrict__ out, int n) {
  int i = blockIdx.x * blockDim.x + threadIdx.x;
  if (i < n) out[i] = (_Float16)in[i];
}

__global__ void add_k(const float* __restrict__ a, const float* __restrict__ b,
                      float* __restrict__ o, int n) {
  int i = blockIdx.x * blockDim.x + threadIdx.x;
  if (i < n) o[i] = a[i] + b[i];
}

// (B,L,C) f16 -> (B*NH, L, HC) f16
__global__ void split_heads_k(const _Float16* __restrict__ in, _Float16* __restrict__ out) {
  int i = blockIdx.x * blockDim.x + threadIdx.x;
  if (i >= BHc * Lc * HCc) return;
  const int hc = i & 31;
  const int m  = (i >> 5) & 1023;
  const int bh = i >> 15;
  const int b  = bh / NHc;
  const int hd = bh - b * NHc;
  out[i] = in[((long long)b * Lc + m) * Cc + hd * HCc + hc];
}

// (B*NH, L, HC) f16 -> (B,L,C) f16
__global__ void merge_heads_k(const _Float16* __restrict__ in, _Float16* __restrict__ out) {
  int i = blockIdx.x * blockDim.x + threadIdx.x;
  if (i >= Bc * Lc * Cc) return;
  const int c = i % Cc;
  const int m = (i / Cc) % Lc;
  const int b = i / (Cc * Lc);
  const int hd = c / HCc, hc = c - hd * HCc;
  out[i] = in[(((long long)(b * NHc + hd)) * Lc + m) * HCc + hc];
}

// (B,NS,C) f16 -> (B*NH, HC, NS) f16   (contraction dim contiguous for attn@V)
__global__ void transpose_v_k(const _Float16* __restrict__ in, _Float16* __restrict__ out) {
  int i = blockIdx.x * blockDim.x + threadIdx.x;
  if (i >= BHc * HCc * NSc) return;
  const int n  = i & 1023;
  const int hc = (i >> 10) & 31;
  const int bh = i >> 15;
  const int b  = bh / NHc;
  const int hd = bh - b * NHc;
  out[i] = in[((long long)b * NSc + n) * Cc + hd * HCc + hc];
}

// ---------------------------------------------------------------------------
// LayerNorm (optionally with residual add): row = (b,l), reduce over C.
//   res_out = x + res (if res given), normf/normh = LN(x+res)*w + b
// ---------------------------------------------------------------------------
__global__ void ln_k(const float* __restrict__ x, const float* __restrict__ res,
                     const float* __restrict__ w, const float* __restrict__ bvec,
                     float* __restrict__ res_out,
                     float* __restrict__ normf, _Float16* __restrict__ normh, int Cdim)
{
  __shared__ float sb[128];
  const int row = blockIdx.x;
  const int tid = threadIdx.x;
  const float* xr = x + (long long)row * Cdim;
  const float* rr = res ? res + (long long)row * Cdim : nullptr;

  float s = 0.f;
  for (int c = tid; c < Cdim; c += blockDim.x) {
    float v = xr[c] + (rr ? rr[c] : 0.f);
    s += v;
  }
  sb[tid] = s; __syncthreads();
  for (int o = 64; o > 0; o >>= 1) { if (tid < o) sb[tid] += sb[tid + o]; __syncthreads(); }
  const float mu = sb[0] / Cdim;
  __syncthreads();

  float s2 = 0.f;
  for (int c = tid; c < Cdim; c += blockDim.x) {
    float v = xr[c] + (rr ? rr[c] : 0.f) - mu;
    s2 += v * v;
  }
  sb[tid] = s2; __syncthreads();
  for (int o = 64; o > 0; o >>= 1) { if (tid < o) sb[tid] += sb[tid + o]; __syncthreads(); }
  const float inv = rsqrtf(sb[0] / Cdim + EPSc);

  for (int c = tid; c < Cdim; c += blockDim.x) {
    float v = xr[c] + (rr ? rr[c] : 0.f);
    if (res_out) res_out[(long long)row * Cdim + c] = v;
    float nv = (v - mu) * inv * w[c] + bvec[c];
    if (normf) normf[(long long)row * Cdim + c] = nv;
    if (normh) normh[(long long)row * Cdim + c] = (_Float16)nv;
  }
}

// ---------------------------------------------------------------------------
// Offset network part 1: depthwise 7x7 conv on q regrouped (B*NG, CG, H, W).
// q is stored (B, L, C) f32; group g covers channels [g*CG, (g+1)*CG).
// ---------------------------------------------------------------------------
__global__ void dwconv7_k(const float* __restrict__ q, const float* __restrict__ wdw,
                          const float* __restrict__ bdw, float* __restrict__ o)
{
  const int idx = blockIdx.x * blockDim.x + threadIdx.x;
  if (idx >= BGc * CGc * Hc * Wc) return;
  const int w  = idx & 31;
  const int h  = (idx >> 5) & 31;
  const int cc = (idx >> 10) & 63;
  const int bg = idx >> 16;
  const int b  = bg / NGc;
  const int g  = bg - b * NGc;
  const int c  = g * CGc + cc;
  const float* wf = wdw + cc * 49;
  float acc = bdw[cc];
  for (int i = 0; i < 7; ++i) {
    const int hh = h + i - 3;
    if (hh < 0 || hh >= Hc) continue;
    for (int j = 0; j < 7; ++j) {
      const int ww = w + j - 3;
      if (ww < 0 || ww >= Wc) continue;
      acc += q[((long long)b * Lc + hh * Wc + ww) * Cc + c] * wf[i * 7 + j];
    }
  }
  o[idx] = acc;   // layout (BG, CG, H*W)
}

// Part 2: LN over CG, GELU, pointwise -> 2, tanh * (1/dim) * ORF + ref grid.
__global__ void offset_head_k(const float* __restrict__ o, const float* __restrict__ lnw,
                              const float* __restrict__ lnb, const float* __restrict__ pw,
                              float* __restrict__ pos)
{
  const int idx = blockIdx.x * blockDim.x + threadIdx.x;
  if (idx >= BGc * Lc) return;
  const int hw = idx & 1023;
  const int bg = idx >> 10;
  float v[CGc];
  float s = 0.f;
  for (int cc = 0; cc < CGc; ++cc) {
    v[cc] = o[((long long)bg * CGc + cc) * Lc + hw];
    s += v[cc];
  }
  const float mu = s * (1.0f / CGc);
  float s2 = 0.f;
  for (int cc = 0; cc < CGc; ++cc) { const float d = v[cc] - mu; s2 += d * d; }
  const float inv = rsqrtf(s2 * (1.0f / CGc) + EPSc);
  float d0 = 0.f, d1 = 0.f;
  for (int cc = 0; cc < CGc; ++cc) {
    float t = (v[cc] - mu) * inv * lnw[cc] + lnb[cc];
    t = gelu_f(t);
    d0 += pw[cc] * t;         // p = 0 (y)
    d1 += pw[CGc + cc] * t;   // p = 1 (x)
  }
  const float offy = tanhf(d0) * (1.0f / Hc) * 2.0f;   // ORF = 2
  const float offx = tanhf(d1) * (1.0f / Wc) * 2.0f;
  const int h = hw >> 5, w = hw & 31;
  pos[idx * 2 + 0] = offy + ((h + 0.5f) * (2.0f / Hc) - 1.0f);
  pos[idx * 2 + 1] = offx + ((w + 0.5f) * (2.0f / Wc) - 1.0f);
}

// Bilinear grid_sample of normed features at pos -> xs (B, NS, C) f16.
// align_corners=True, zero padding; grid x = pos[...,1], y = pos[...,0].
__global__ void sample_xs_k(const float* __restrict__ xn, const float* __restrict__ pos,
                            _Float16* __restrict__ xs)
{
  const int idx = blockIdx.x * blockDim.x + threadIdx.x;
  if (idx >= BGc * CGc * NSc) return;
  const int n  = idx & 1023;
  const int cc = (idx >> 10) & 63;
  const int bg = idx >> 16;
  const int b  = bg / NGc;
  const int g  = bg - b * NGc;
  const int c  = g * CGc + cc;
  const float py = pos[(bg * NSc + n) * 2 + 0];
  const float px = pos[(bg * NSc + n) * 2 + 1];
  const float fx = (px + 1.0f) * 0.5f * (Wc - 1);
  const float fy = (py + 1.0f) * 0.5f * (Hc - 1);
  const float x0f = floorf(fx), y0f = floorf(fy);
  const int x0 = (int)x0f, y0 = (int)y0f;
  const float wx = fx - x0f, wy = fy - y0f;
  auto at = [&](int xi, int yi) -> float {
    if (xi < 0 || xi >= Wc || yi < 0 || yi >= Hc) return 0.f;
    return xn[((long long)b * Lc + yi * Wc + xi) * Cc + c];
  };
  float acc = at(x0, y0) * (1.f - wx) * (1.f - wy)
            + at(x0 + 1, y0) * wx * (1.f - wy)
            + at(x0, y0 + 1) * (1.f - wx) * wy
            + at(x0 + 1, y0 + 1) * wx * wy;
  xs[((long long)b * NSc + n) * Cc + c] = (_Float16)acc;
}

// ---------------------------------------------------------------------------
// Fused: attn = softmax(attn_raw * scale + rpe_bias) over n, in place (f16).
// One 256-thread block per row (bh, m); 4 columns per thread held in regs.
// ---------------------------------------------------------------------------
__device__ __forceinline__ float rpe_at(const float* __restrict__ rp, int xi, int yi) {
  if (xi < 0 || xi > 62 || yi < 0 || yi > 62) return 0.f;
  return rp[yi * 63 + xi];
}

__global__ void bias_softmax_k(_Float16* __restrict__ attn,
                               const float* __restrict__ pos,
                               const float* __restrict__ rpe,
                               float scale)
{
  __shared__ float sb[256];
  const int row = blockIdx.x;          // bh*1024 + m
  const int m  = row & 1023;
  const int bh = row >> 10;
  const int b  = bh / NHc;
  const int hd = bh - b * NHc;
  const int g  = hd / GHc;
  const int bg = b * NGc + g;
  const int mh = m >> 5, mw = m & 31;
  const float qy = (mh + 0.5f) * (2.0f / Hc) - 1.0f;
  const float qx = (mw + 0.5f) * (2.0f / Wc) - 1.0f;
  const float* rp = rpe + hd * 63 * 63;
  _Float16* arow = attn + (long long)row * NSc;
  const int tid = threadIdx.x;

  float vals[4];
  float mx = -3.0e38f;
#pragma unroll
  for (int i = 0; i < 4; ++i) {
    const int n = tid + i * 256;
    const float py = pos[(bg * NSc + n) * 2 + 0];
    const float px = pos[(bg * NSc + n) * 2 + 1];
    const float dx = (qx - px) * 0.5f;
    const float dy = (qy - py) * 0.5f;
    const float fx = (dx + 1.0f) * 0.5f * 62.0f;
    const float fy = (dy + 1.0f) * 0.5f * 62.0f;
    const float x0f = floorf(fx), y0f = floorf(fy);
    const int x0 = (int)x0f, y0 = (int)y0f;
    const float wx = fx - x0f, wy = fy - y0f;
    float bv = rpe_at(rp, x0, y0) * (1.f - wx) * (1.f - wy)
             + rpe_at(rp, x0 + 1, y0) * wx * (1.f - wy)
             + rpe_at(rp, x0, y0 + 1) * (1.f - wx) * wy
             + rpe_at(rp, x0 + 1, y0 + 1) * wx * wy;
    float v = (float)arow[n] * scale + bv;
    vals[i] = v;
    mx = fmaxf(mx, v);
  }
  sb[tid] = mx; __syncthreads();
  for (int o = 128; o > 0; o >>= 1) { if (tid < o) sb[tid] = fmaxf(sb[tid], sb[tid + o]); __syncthreads(); }
  mx = sb[0]; __syncthreads();

  float se = 0.f;
#pragma unroll
  for (int i = 0; i < 4; ++i) { vals[i] = expf(vals[i] - mx); se += vals[i]; }
  sb[tid] = se; __syncthreads();
  for (int o = 128; o > 0; o >>= 1) { if (tid < o) sb[tid] += sb[tid + o]; __syncthreads(); }
  const float rinv = 1.0f / sb[0];
  __syncthreads();
#pragma unroll
  for (int i = 0; i < 4; ++i) arow[tid + i * 256] = (_Float16)(vals[i] * rinv);
}

// ---------------------------------------------------------------------------
// Host side
// ---------------------------------------------------------------------------
extern "C" void kernel_launch(void* const* d_in, const int* in_sizes, int n_in,
                              void* d_out, int out_size, void* d_ws, size_t ws_size,
                              hipStream_t stream)
{
  (void)in_sizes; (void)n_in; (void)out_size; (void)ws_size;
  const float* x    = (const float*)d_in[0];
  const float* n1w  = (const float*)d_in[1];
  const float* n1b  = (const float*)d_in[2];
  const float* wq   = (const float*)d_in[3];
  const float* bq   = (const float*)d_in[4];
  const float* wk   = (const float*)d_in[5];
  const float* bk   = (const float*)d_in[6];
  const float* wv   = (const float*)d_in[7];
  const float* bv   = (const float*)d_in[8];
  const float* wo   = (const float*)d_in[9];
  const float* bo   = (const float*)d_in[10];
  const float* dww  = (const float*)d_in[11];
  const float* dwb  = (const float*)d_in[12];
  const float* olnw = (const float*)d_in[13];
  const float* olnb = (const float*)d_in[14];
  const float* opw  = (const float*)d_in[15];
  const float* rpe  = (const float*)d_in[16];
  const float* n2w  = (const float*)d_in[17];
  const float* n2b  = (const float*)d_in[18];
  const float* fc1w = (const float*)d_in[19];
  const float* fc1b = (const float*)d_in[20];
  const float* fc2w = (const float*)d_in[21];
  const float* fc2b = (const float*)d_in[22];
  float* out = (float*)d_out;

  char* ws = (char*)d_ws;
  size_t off = 0;
  auto alloc = [&](size_t bytes) -> void* {
    void* p = ws + off;
    off += (bytes + 255) & ~(size_t)255;
    return p;
  };

  const int ML = Bc * Lc;                          // 2048 rows
  float*    xnf   = (float*)   alloc((size_t)ML * Cc * 4);
  _Float16* xnh   = (_Float16*)alloc((size_t)ML * Cc * 2);
  _Float16* wqh   = (_Float16*)alloc((size_t)Cc * Cc * 2);
  _Float16* wkh   = (_Float16*)alloc((size_t)Cc * Cc * 2);
  _Float16* wvh   = (_Float16*)alloc((size_t)Cc * Cc * 2);
  _Float16* woh   = (_Float16*)alloc((size_t)Cc * Cc * 2);
  _Float16* fc1wh = (_Float16*)alloc((size_t)HIDc * Cc * 2);
  _Float16* fc2wh = (_Float16*)alloc((size_t)Cc * HIDc * 2);
  float*    qf    = (float*)   alloc((size_t)ML * Cc * 4);
  _Float16* qh    = (_Float16*)alloc((size_t)ML * Cc * 2);
  float*    odw   = (float*)   alloc((size_t)BGc * CGc * Lc * 4);
  float*    pos   = (float*)   alloc((size_t)BGc * NSc * 2 * 4);
  _Float16* xsh   = (_Float16*)alloc((size_t)ML * Cc * 2);
  _Float16* kh    = (_Float16*)alloc((size_t)ML * Cc * 2);
  _Float16* vh    = (_Float16*)alloc((size_t)ML * Cc * 2);
  _Float16* qsh   = (_Float16*)alloc((size_t)BHc * Lc * HCc * 2);
  _Float16* ksh   = (_Float16*)alloc((size_t)BHc * NSc * HCc * 2);
  _Float16* vT    = (_Float16*)alloc((size_t)BHc * HCc * NSc * 2);
  _Float16* attnS = (_Float16*)alloc((size_t)BHc * Lc * NSc * 2);   // 50 MB
  _Float16* avh   = (_Float16*)alloc((size_t)BHc * Lc * HCc * 2);
  _Float16* oh    = (_Float16*)alloc((size_t)ML * Cc * 2);
  float*    yf    = (float*)   alloc((size_t)ML * Cc * 4);
  float*    x2f   = (float*)   alloc((size_t)ML * Cc * 4);
  _Float16* hnh   = (_Float16*)alloc((size_t)ML * Cc * 2);
  _Float16* h1h   = (_Float16*)alloc((size_t)ML * HIDc * 2);
  float*    y2f   = (float*)   alloc((size_t)ML * Cc * 4);

  const dim3 gblk(32, 4, 1);
  auto gemm_grid = [](int M, int N, int batch) {
    int tiles = (M / 32) * (N / 32);
    return dim3((tiles + 3) / 4, 1, batch);
  };
  auto ceil256 = [](int n) { return (n + 255) / 256; };

  // ---- weight conversions to f16 ----
  to_half_k<<<ceil256(Cc * Cc), 256, 0, stream>>>(wq, wqh, Cc * Cc);
  to_half_k<<<ceil256(Cc * Cc), 256, 0, stream>>>(wk, wkh, Cc * Cc);
  to_half_k<<<ceil256(Cc * Cc), 256, 0, stream>>>(wv, wvh, Cc * Cc);
  to_half_k<<<ceil256(Cc * Cc), 256, 0, stream>>>(wo, woh, Cc * Cc);
  to_half_k<<<ceil256(HIDc * Cc), 256, 0, stream>>>(fc1w, fc1wh, HIDc * Cc);
  to_half_k<<<ceil256(Cc * HIDc), 256, 0, stream>>>(fc2w, fc2wh, Cc * HIDc);

  // ---- LN1 ----
  ln_k<<<ML, 128, 0, stream>>>(x, nullptr, n1w, n1b, nullptr, xnf, xnh, Cc);

  // ---- Q projection (f32 for offset net + f16 for attention) ----
  wmma_gemm_nt<<<gemm_grid(ML, Cc, 1), gblk, 0, stream>>>(
      xnh, Cc, 0LL, wqh, Cc, 0LL, qf, qh, Cc, 0LL, bq, ML, Cc, Cc, 1.0f, 0);

  // ---- offset network -> pos ----
  dwconv7_k<<<ceil256(BGc * CGc * Lc), 256, 0, stream>>>(qf, dww, dwb, odw);
  offset_head_k<<<ceil256(BGc * Lc), 256, 0, stream>>>(odw, olnw, olnb, opw, pos);

  // ---- deformable sampling of normed features ----
  sample_xs_k<<<ceil256(BGc * CGc * NSc), 256, 0, stream>>>(xnf, pos, xsh);

  // ---- K / V projections (f16 only) ----
  wmma_gemm_nt<<<gemm_grid(ML, Cc, 1), gblk, 0, stream>>>(
      xsh, Cc, 0LL, wkh, Cc, 0LL, nullptr, kh, Cc, 0LL, bk, ML, Cc, Cc, 1.0f, 0);
  wmma_gemm_nt<<<gemm_grid(ML, Cc, 1), gblk, 0, stream>>>(
      xsh, Cc, 0LL, wvh, Cc, 0LL, nullptr, vh, Cc, 0LL, bv, ML, Cc, Cc, 1.0f, 0);

  // ---- head reshapes ----
  split_heads_k<<<ceil256(BHc * Lc * HCc), 256, 0, stream>>>(qh, qsh);
  split_heads_k<<<ceil256(BHc * NSc * HCc), 256, 0, stream>>>(kh, ksh);
  transpose_v_k<<<ceil256(BHc * HCc * NSc), 256, 0, stream>>>(vh, vT);

  // ---- attention scores: (BH) x [1024 x 1024] = Q(1024x32) . K(1024x32)^T ----
  wmma_gemm_nt<<<gemm_grid(Lc, NSc, BHc), gblk, 0, stream>>>(
      qsh, HCc, (long long)Lc * HCc,
      ksh, HCc, (long long)NSc * HCc,
      nullptr, attnS, NSc, (long long)Lc * NSc,
      nullptr, Lc, NSc, HCc, 1.0f, 0);

  // ---- fused scale + continuous RPE bias + softmax (in place, f16) ----
  bias_softmax_k<<<BHc * Lc, 256, 0, stream>>>(attnS, pos, rpe, 0.17677669529663687f);

  // ---- attn @ V: (BH) x [1024 x 32] ----
  wmma_gemm_nt<<<gemm_grid(Lc, HCc, BHc), gblk, 0, stream>>>(
      attnS, NSc, (long long)Lc * NSc,
      vT, NSc, (long long)HCc * NSc,
      nullptr, avh, HCc, (long long)Lc * HCc,
      nullptr, Lc, HCc, NSc, 1.0f, 0);

  merge_heads_k<<<ceil256(Bc * Lc * Cc), 256, 0, stream>>>(avh, oh);

  // ---- output projection ----
  wmma_gemm_nt<<<gemm_grid(ML, Cc, 1), gblk, 0, stream>>>(
      oh, Cc, 0LL, woh, Cc, 0LL, yf, nullptr, Cc, 0LL, bo, ML, Cc, Cc, 1.0f, 0);

  // ---- residual + LN2 ----
  ln_k<<<ML, 128, 0, stream>>>(x, yf, n2w, n2b, x2f, nullptr, hnh, Cc);

  // ---- MLP ----
  wmma_gemm_nt<<<gemm_grid(ML, HIDc, 1), gblk, 0, stream>>>(
      hnh, Cc, 0LL, fc1wh, Cc, 0LL, nullptr, h1h, HIDc, 0LL, fc1b, ML, HIDc, Cc, 1.0f, 1);
  wmma_gemm_nt<<<gemm_grid(ML, Cc, 1), gblk, 0, stream>>>(
      h1h, HIDc, 0LL, fc2wh, HIDc, 0LL, y2f, nullptr, Cc, 0LL, fc2b, ML, Cc, HIDc, 1.0f, 0);

  // ---- final residual ----
  add_k<<<ceil256(ML * Cc), 256, 0, stream>>>(x2f, y2f, out, ML * Cc);
}
